// TensorTrainLinear_82111184764954
// MI455X (gfx1250) — matrix operand order repaired and checked by reading, not compiled, not run on hardware
//
#include <hip/hip_runtime.h>
#include <hip/hip_bf16.h>
#include <stdint.h>

// ---------------------------------------------------------------------------
// TT-linear on MI455X: materialize dense W (bf16, L2-resident), then one
// bf16 WMMA GEMM: out[2048,4096] = x[2048,4096] @ W[4096,4096] + bias.
// Tile feed uses gfx1250 GLOBAL_LOAD_ASYNC_TO_LDS (ASYNCcnt, confirmed to
// lower); triple-buffered LDS, one s_wait_asynccnt + one barrier per K-iter.
// Wave tile 64x64: 16 wmma per 16 ds_load_b128 (A/B frags each reused 4x).
// ---------------------------------------------------------------------------

typedef __attribute__((ext_vector_type(16))) __bf16       v16bf;
typedef __attribute__((ext_vector_type(8)))  float        v8f;
typedef __attribute__((ext_vector_type(4)))  unsigned int u32x4;
typedef __attribute__((ext_vector_type(2)))  unsigned int u32x2;
typedef __attribute__((ext_vector_type(4)))  float        f32x4;
typedef int v4i __attribute__((vector_size(16)));   // matches builtin param type

#define IN_F  4096
#define OUT_F 4096

#if defined(__AMDGCN__) && \
    __has_builtin(__builtin_amdgcn_global_load_async_to_lds_b128) && \
    __has_builtin(__builtin_amdgcn_s_wait_asynccnt)
#define USE_ASYNC_LDS 1
#else
#define USE_ASYNC_LDS 0
#endif

__device__ __forceinline__ unsigned short f2bf(float f) {
  unsigned u = __builtin_bit_cast(unsigned, f);
  u += 0x7FFFu + ((u >> 16) & 1u);          // round-to-nearest-even
  return (unsigned short)(u >> 16);
}

union FragBF {
  v16bf v;
  u32x4 q[2];
};

#if USE_ASYNC_LDS
__device__ __forceinline__ void async_cp16(const unsigned short* g,
                                           unsigned short* l) {
  __builtin_amdgcn_global_load_async_to_lds_b128(
      (__attribute__((address_space(1))) v4i*)(uintptr_t)g,
      (__attribute__((address_space(3))) v4i*)(unsigned)(uintptr_t)l,
      0, 0);
}
#endif

// ---------------------------------------------------------------------------
// Kernel 1: C12[q][i2][o2][i3][o3] = sum_r core1[q][i2][o2][r]*core2[r][i3][o3]
// ---------------------------------------------------------------------------
__global__ void k_c12(const float* __restrict__ core1,
                      const float* __restrict__ core2,
                      float* __restrict__ c12) {
  int idx = blockIdx.x * 256 + threadIdx.x;
  int o3 = idx & 15;
  int i3 = (idx >> 4) & 15;
  int o2 = (idx >> 8) & 15;
  int i2 = (idx >> 12) & 15;
  int q  = idx >> 16;
  const float* c1 = core1 + (((q * 16 + i2) * 16 + o2) * 16);
  const float* c2 = core2 + (i3 * 16 + o3);
  float acc = 0.f;
#pragma unroll
  for (int r = 0; r < 16; ++r) acc += c1[r] * c2[r * 256];
  c12[idx] = acc;
}

// ---------------------------------------------------------------------------
// Kernel 2: Wt[(o1,o2,o3)][(i1,i2,i3)] (bf16, N-major i.e. W transposed)
//   W[(i1,i2,i3),(o1,o2,o3)] = sum_q core0[i1][o1][q] * C12[q][i2][o2][i3][o3]
// ---------------------------------------------------------------------------
__global__ void k_wt(const float* __restrict__ core0,
                     const float* __restrict__ c12,
                     unsigned short* __restrict__ wt) {
  __shared__ float s[16 * 256];
  __shared__ float s0[16];
  int bb = blockIdx.x;
  int o2 = bb & 15;
  int o1 = (bb >> 4) & 15;
  int i2 = (bb >> 8) & 15;
  int i1 = bb >> 12;
  int tid = threadIdx.x;
  int base = i2 * 4096 + o2 * 256;
#pragma unroll
  for (int q = 0; q < 16; ++q) s[q * 256 + tid] = c12[q * 65536 + base + tid];
  if (tid < 16) s0[tid] = core0[(i1 * 16 + o1) * 16 + tid];
  __syncthreads();
  int i3 = tid & 15;
  int o3 = tid >> 4;
  int li = i3 * 16 + o3;
  float acc = 0.f;
#pragma unroll
  for (int q = 0; q < 16; ++q) acc += s0[q] * s[q * 256 + li];
  wt[(size_t)(o1 * 256 + o2 * 16 + o3) * IN_F + (i1 * 256 + i2 * 16 + i3)] = f2bf(acc);
}

// ---------------------------------------------------------------------------
// Kernel 3: x fp32 -> bf16
// ---------------------------------------------------------------------------
__global__ void k_xcast(const float* __restrict__ x,
                        unsigned short* __restrict__ xb, int n4) {
  int i = blockIdx.x * 256 + threadIdx.x;
  if (i >= n4) return;
  f32x4 v = *(const f32x4*)(x + (size_t)i * 4);
  u32x2 p;
  p.x = (unsigned)f2bf(v.x) | ((unsigned)f2bf(v.y) << 16);
  p.y = (unsigned)f2bf(v.z) | ((unsigned)f2bf(v.w) << 16);
  *(u32x2*)(xb + (size_t)i * 4) = p;
}

// ---------------------------------------------------------------------------
// Kernel 4: bf16 WMMA GEMM, block tile 128x256, BK=32, wave tile 64x64.
// ---------------------------------------------------------------------------
#define BM 128
#define BN 256
#define BK 32
#define LDT 40
#define KTILES (IN_F / BK)

__global__ __launch_bounds__(256) void k_gemm(
    const unsigned short* __restrict__ Xb,   // [M][4096] bf16
    const unsigned short* __restrict__ Wt,   // [4096 N][4096 K] bf16
    const float* __restrict__ bias,
    float* __restrict__ out) {
#if USE_ASYNC_LDS
  __shared__ unsigned short sA[3][BM * LDT];
  __shared__ unsigned short sB[3][BN * LDT];
#else
  __shared__ unsigned short sA[2][BM * LDT];
  __shared__ unsigned short sB[2][BN * LDT];
#endif

  const int tid  = threadIdx.x;
  const int lane = tid & 31;
  const int wave = tid >> 5;
  const int wm   = wave & 1;    // 0..1 over M (64 rows each)
  const int wn   = wave >> 1;   // 0..3 over N (64 cols each)
  const int lr   = lane & 15;
  const int lh   = lane >> 4;

  const int row0 = blockIdx.y * BM;
  const int col0 = blockIdx.x * BN;

  // global->LDS staging: A 2 chunks/thread, B 4 chunks/thread (16B each)
  const int r0 = tid >> 2;   // 0..63 : tile row
  const int kc = tid & 3;    // 16B chunk within the 64B (32-elem) K-slice

  const unsigned short* aG = Xb + (size_t)(row0 + r0) * IN_F + kc * 8;
  const unsigned short* bG = Wt + (size_t)(col0 + r0) * IN_F + kc * 8;
  const size_t rstep = (size_t)64 * IN_F;

  v8f acc[4][4];
  v8f vz = {};
#pragma unroll
  for (int i = 0; i < 4; ++i)
#pragma unroll
    for (int j = 0; j < 4; ++j) acc[i][j] = vz;

#if USE_ASYNC_LDS
  auto issue = [&](int kt, int bf) {
    const size_t ko = (size_t)kt * BK;
    async_cp16(aG + ko,             &sA[bf][r0 * LDT + kc * 8]);
    async_cp16(aG + rstep + ko,     &sA[bf][(r0 + 64) * LDT + kc * 8]);
    async_cp16(bG + ko,             &sB[bf][r0 * LDT + kc * 8]);
    async_cp16(bG + rstep + ko,     &sB[bf][(r0 + 64) * LDT + kc * 8]);
    async_cp16(bG + 2 * rstep + ko, &sB[bf][(r0 + 128) * LDT + kc * 8]);
    async_cp16(bG + 3 * rstep + ko, &sB[bf][(r0 + 192) * LDT + kc * 8]);
  };
  issue(0, 0);
  issue(1, 1);

  for (int kt = 0; kt < KTILES; ++kt) {
    const int buf = kt % 3;
    // own stage-kt copies landed (in-order completion; stage kt+1 may remain)
    if (kt + 1 < KTILES) __builtin_amdgcn_s_wait_asynccnt(6);
    else                 __builtin_amdgcn_s_wait_asynccnt(0);
    __syncthreads();   // all waves' stage-kt LDS writes visible
    // buffer (kt+2)%3 was last read at compute(kt-1), finished before this
    // barrier by every wave -> safe to overwrite now.
    if (kt + 2 < KTILES) issue(kt + 2, (kt + 2) % 3);

    FragBF a[4], b[4];
#pragma unroll
    for (int mt = 0; mt < 4; ++mt) {
      const int row = wm * 64 + mt * 16 + lr;
      a[mt].q[0] = *(const u32x4*)&sA[buf][row * LDT + lh * 8];
      a[mt].q[1] = *(const u32x4*)&sA[buf][row * LDT + 16 + lh * 8];
    }
#pragma unroll
    for (int nt = 0; nt < 4; ++nt) {
      const int col = wn * 64 + nt * 16 + lr;
      b[nt].q[0] = *(const u32x4*)&sB[buf][col * LDT + lh * 8];
      b[nt].q[1] = *(const u32x4*)&sB[buf][col * LDT + 16 + lh * 8];
    }
#pragma unroll
    for (int mt = 0; mt < 4; ++mt)
#pragma unroll
      for (int nt = 0; nt < 4; ++nt)
        acc[mt][nt] = __builtin_amdgcn_wmma_f32_16x16x32_bf16(
            false, a[mt].v, false, b[nt].v, (short)0, acc[mt][nt], false, false);
  }
#else  // !USE_ASYNC_LDS : register-staged double buffer
  u32x4 ar0 = *(const u32x4*)(aG);
  u32x4 ar1 = *(const u32x4*)(aG + rstep);
  u32x4 br0 = *(const u32x4*)(bG);
  u32x4 br1 = *(const u32x4*)(bG + rstep);
  u32x4 br2 = *(const u32x4*)(bG + 2 * rstep);
  u32x4 br3 = *(const u32x4*)(bG + 3 * rstep);

  *(u32x4*)&sA[0][r0 * LDT + kc * 8]          = ar0;
  *(u32x4*)&sA[0][(r0 + 64) * LDT + kc * 8]   = ar1;
  *(u32x4*)&sB[0][r0 * LDT + kc * 8]          = br0;
  *(u32x4*)&sB[0][(r0 + 64) * LDT + kc * 8]   = br1;
  *(u32x4*)&sB[0][(r0 + 128) * LDT + kc * 8]  = br2;
  *(u32x4*)&sB[0][(r0 + 192) * LDT + kc * 8]  = br3;
  __syncthreads();

  for (int kt = 0; kt < KTILES; ++kt) {
    const int buf = kt & 1;
    if (kt + 1 < KTILES) {
      const size_t ko = (size_t)(kt + 1) * BK;
      ar0 = *(const u32x4*)(aG + ko);
      ar1 = *(const u32x4*)(aG + rstep + ko);
      br0 = *(const u32x4*)(bG + ko);
      br1 = *(const u32x4*)(bG + rstep + ko);
      br2 = *(const u32x4*)(bG + 2 * rstep + ko);
      br3 = *(const u32x4*)(bG + 3 * rstep + ko);
    }

    FragBF a[4], b[4];
#pragma unroll
    for (int mt = 0; mt < 4; ++mt) {
      const int row = wm * 64 + mt * 16 + lr;
      a[mt].q[0] = *(const u32x4*)&sA[buf][row * LDT + lh * 8];
      a[mt].q[1] = *(const u32x4*)&sA[buf][row * LDT + 16 + lh * 8];
    }
#pragma unroll
    for (int nt = 0; nt < 4; ++nt) {
      const int col = wn * 64 + nt * 16 + lr;
      b[nt].q[0] = *(const u32x4*)&sB[buf][col * LDT + lh * 8];
      b[nt].q[1] = *(const u32x4*)&sB[buf][col * LDT + 16 + lh * 8];
    }
#pragma unroll
    for (int mt = 0; mt < 4; ++mt)
#pragma unroll
      for (int nt = 0; nt < 4; ++nt)
        acc[mt][nt] = __builtin_amdgcn_wmma_f32_16x16x32_bf16(
            false, a[mt].v, false, b[nt].v, (short)0, acc[mt][nt], false, false);

    if (kt + 1 < KTILES) {
      const int nb = buf ^ 1;
      *(u32x4*)&sA[nb][r0 * LDT + kc * 8]          = ar0;
      *(u32x4*)&sA[nb][(r0 + 64) * LDT + kc * 8]   = ar1;
      *(u32x4*)&sB[nb][r0 * LDT + kc * 8]          = br0;
      *(u32x4*)&sB[nb][(r0 + 64) * LDT + kc * 8]   = br1;
      *(u32x4*)&sB[nb][(r0 + 128) * LDT + kc * 8]  = br2;
      *(u32x4*)&sB[nb][(r0 + 192) * LDT + kc * 8]  = br3;
      __syncthreads();
    }
  }
#endif

  // Epilogue: C/D layout -> n = lane&15, m = r + 8*(lane>>4)
#pragma unroll
  for (int nt = 0; nt < 4; ++nt) {
    const int n = col0 + wn * 64 + nt * 16 + lr;
    const float bv = bias[n];
#pragma unroll
    for (int mt = 0; mt < 4; ++mt) {
      const int mbase = row0 + wm * 64 + mt * 16 + lh * 8;
#pragma unroll
      for (int r = 0; r < 8; ++r)
        out[(size_t)(mbase + r) * OUT_F + n] = acc[mt][nt][r] + bv;
    }
  }
}

// ---------------------------------------------------------------------------
extern "C" void kernel_launch(void* const* d_in, const int* in_sizes, int n_in,
                              void* d_out, int out_size, void* d_ws, size_t ws_size,
                              hipStream_t stream) {
  (void)n_in; (void)out_size; (void)ws_size;
  const float* x     = (const float*)d_in[0];
  const float* core0 = (const float*)d_in[1];
  const float* core1 = (const float*)d_in[2];
  const float* core2 = (const float*)d_in[3];
  const float* bias  = (const float*)d_in[4];
  float* out = (float*)d_out;

  const int rows = in_sizes[0] / IN_F;   // 2048

  // workspace layout: [ Wt bf16 : 33.5MB ][ C12 fp32 (4MB) / Xb bf16 (16.8MB) ]
  // C12 is dead after k_wt; stream ordering makes the aliasing safe.
  char* ws = (char*)d_ws;
  unsigned short* wt  = (unsigned short*)ws;
  char*           ws2 = ws + (size_t)IN_F * OUT_F * 2;
  float*          c12 = (float*)ws2;
  unsigned short* xb  = (unsigned short*)ws2;

  k_c12<<<4096, 256, 0, stream>>>(core1, core2, c12);
  k_wt<<<65536, 256, 0, stream>>>(core0, c12, wt);

  const int n4 = rows * IN_F / 4;
  k_xcast<<<(n4 + 255) / 256, 256, 0, stream>>>(x, xb, n4);

  dim3 grid(OUT_F / BN, rows / BM);
  k_gemm<<<grid, 256, 0, stream>>>(xb, wt, bias, out);
}
